// GraphModel1_352187318739
// MI455X (gfx1250) — matrix operand order repaired and checked by reading, not compiled, not run on hardware
//
#include <hip/hip_runtime.h>
#include <hip/hip_bf16.h>
#include <math.h>

#define NNODES 20000
#define NEDGES 320000
#define NGRAPH 32
#define NHEADS 5
#define DHEAD  32
#define FOUT   160      // H * DHEAD
#define FQW    640      // Q|K|V|S fused width
#define EDIM   8
#define NTILES 40       // FQW / 16

typedef __attribute__((ext_vector_type(16))) __bf16 v16bf;
typedef __attribute__((ext_vector_type(8)))  __bf16 v8bf;
typedef __attribute__((ext_vector_type(8)))  float  v8f;

// ---------------------------------------------------------------- utilities
__device__ inline void atomicMaxF(float* addr, float val) {
    float old = *addr;
    while (val > old) {
        int assumed = __float_as_int(old);
        int prev = atomicCAS((int*)addr, assumed, __float_as_int(val));
        if (prev == assumed) break;
        old = __int_as_float(prev);
    }
}

// --------------------------------------------------- f32 -> bf16 (padded K)
__global__ void k_cvt_bf16(const float* __restrict__ src, __bf16* __restrict__ dstb,
                           int n, int din, int dinP) {
    int i = blockIdx.x * blockDim.x + threadIdx.x;
    if (i >= n * dinP) return;
    int c = i % dinP, r = i / dinP;
    float v = (c < din) ? src[(size_t)r * din + c] : 0.0f;
    dstb[i] = (__bf16)v;
}

// ------------------------------------- pack W[q|k|v|s] into B-fragment tiles
// tile(kt,nt): 32 lanes x 16 bf16; lane (h=lane>>4, col=nt*16+(lane&15))
// holds K = kt*32 + h*8 + (j&7) + (j>>3)*16  (j = 0..15), matching wave32 B layout.
__global__ void k_pack_w(const float* __restrict__ Wq, const float* __restrict__ Wk,
                         const float* __restrict__ Wv, const float* __restrict__ Ws,
                         __bf16* __restrict__ BP, int din, int kTiles) {
    int i = blockIdx.x * blockDim.x + threadIdx.x;
    if (i >= kTiles * NTILES * 32 * 16) return;
    int j    = i & 15;
    int lane = (i >> 4) & 31;
    int nt   = (i >> 9) % NTILES;
    int kt   = i / (512 * NTILES);
    int col  = nt * 16 + (lane & 15);
    int k    = kt * 32 + (lane >> 4) * 8 + (j & 7) + (j >> 3) * 16;
    const float* W4[4] = {Wq, Wk, Wv, Ws};
    int m = col / FOUT, c = col - m * FOUT;
    float v = (k < din) ? W4[m][(size_t)k * FOUT + c] : 0.0f;
    BP[i] = (__bf16)v;
}

__global__ void k_pack_bias(const float* __restrict__ bq, const float* __restrict__ bk,
                            const float* __restrict__ bv, const float* __restrict__ bs,
                            float* __restrict__ B) {
    int i = blockIdx.x * blockDim.x + threadIdx.x;
    if (i >= FQW) return;
    const float* b4[4] = {bq, bk, bv, bs};
    int m = i / FOUT;
    B[i] = b4[m][i - m * FOUT];
}

// -------------------------------------------- fused QKVS GEMM, bf16 WMMA
// C[M,640] = XB[M,dinP](bf16) @ BP + bias.  4 waves/block, each wave owns one
// 16-row tile and 4 adjacent 16-col tiles (A-fragment reuse x4).
__global__ void __launch_bounds__(128)
k_gemm_qkvs(const __bf16* __restrict__ XB, const __bf16* __restrict__ BP,
            const float* __restrict__ bias, float* __restrict__ C,
            int Mtiles, int dinP) {
    const int lane = threadIdx.x & 31;
    const int wave = threadIdx.x >> 5;
    const int mt = blockIdx.x * 4 + wave;
    if (mt >= Mtiles) return;                    // wave-uniform: EXEC stays all-1
    const int nt0  = blockIdx.y * 4;
    const int half = lane >> 4;
    const int r    = lane & 15;
    const int kTiles = dinP >> 5;

    v8f acc0 = {}, acc1 = {}, acc2 = {}, acc3 = {};
    const __bf16* aBase = XB + (size_t)(mt * 16 + r) * dinP + half * 8;
    const __bf16* bBase = BP + ((size_t)nt0 * 32 + lane) * 16;

    for (int kt = 0; kt < kTiles; ++kt) {
        v8bf lo = *(const v8bf*)(aBase + kt * 32);
        v8bf hi = *(const v8bf*)(aBase + kt * 32 + 16);
        v16bf a;
#pragma unroll
        for (int q = 0; q < 8; ++q) { a[q] = lo[q]; a[q + 8] = hi[q]; }
        const __bf16* bp = bBase + (size_t)kt * NTILES * 512;
        v16bf b0 = *(const v16bf*)(bp);
        v16bf b1 = *(const v16bf*)(bp + 512);
        v16bf b2 = *(const v16bf*)(bp + 1024);
        v16bf b3 = *(const v16bf*)(bp + 1536);
        acc0 = __builtin_amdgcn_wmma_f32_16x16x32_bf16(false, a, false, b0, (short)0, acc0, false, false);
        acc1 = __builtin_amdgcn_wmma_f32_16x16x32_bf16(false, a, false, b1, (short)0, acc1, false, false);
        acc2 = __builtin_amdgcn_wmma_f32_16x16x32_bf16(false, a, false, b2, (short)0, acc2, false, false);
        acc3 = __builtin_amdgcn_wmma_f32_16x16x32_bf16(false, a, false, b3, (short)0, acc3, false, false);
    }
    const int colBase = nt0 * 16 + r;
    float b0v = bias[colBase], b1v = bias[colBase + 16];
    float b2v = bias[colBase + 32], b3v = bias[colBase + 48];
    float* crow = C + (size_t)(mt * 16 + half * 8) * FQW;
#pragma unroll
    for (int j = 0; j < 8; ++j) {
        float* cp = crow + (size_t)j * FQW;
        cp[colBase]      = acc0[j] + b0v;
        cp[colBase + 16] = acc1[j] + b1v;
        cp[colBase + 32] = acc2[j] + b2v;
        cp[colBase + 48] = acc3[j] + b3v;
    }
}

// --------------------------------------------------------- per-layer init
__global__ void k_init_layer(float* __restrict__ ACC, float* __restrict__ MX,
                             float* __restrict__ DEN) {
    int i = blockIdx.x * blockDim.x + threadIdx.x;
    if (i < NNODES * FOUT) ACC[i] = 0.0f;
    if (i < NNODES * NHEADS) { MX[i] = -1e30f; DEN[i] = 0.0f; }
}

// ------------------------------- edge pass 1: alpha + segment max (per dst)
__global__ void __launch_bounds__(256)
k_edge_alpha(const float* __restrict__ FQ, const float* __restrict__ attr,
             const float* __restrict__ We, const int* __restrict__ src,
             const int* __restrict__ dst, float* __restrict__ EX,
             float* __restrict__ MX) {
    __shared__ float sWe[EDIM * FOUT];
    for (int i = threadIdx.x; i < EDIM * FOUT; i += blockDim.x) sWe[i] = We[i];
    __syncthreads();
    int e = blockIdx.x * 8 + (threadIdx.x >> 5);
    if (e >= NEDGES) return;
    int lane = threadIdx.x & 31;
    int s = src[e], d = dst[e];
    float a8[EDIM];
#pragma unroll
    for (int j = 0; j < EDIM; ++j) a8[j] = attr[(size_t)e * EDIM + j];
    const float* qrow = FQ + (size_t)d * FQW;
    const float* krow = FQ + (size_t)s * FQW + FOUT;
#pragma unroll
    for (int h = 0; h < NHEADS; ++h) {
        int c = h * DHEAD + lane;
        float ep = 0.0f;
#pragma unroll
        for (int j = 0; j < EDIM; ++j) ep += a8[j] * sWe[j * FOUT + c];
        float p = qrow[c] * (krow[c] + ep);
#pragma unroll
        for (int off = 16; off > 0; off >>= 1) p += __shfl_xor(p, off, 32);
        if (lane == 0) {
            float al = p * 0.17677669529663687f;   // 1/sqrt(32)
            EX[(size_t)e * NHEADS + h] = al;
            atomicMaxF(&MX[d * NHEADS + h], al);
        }
    }
}

// ------------------------------- edge pass 2: exp + segment sum of denom
__global__ void k_edge_exp(float* __restrict__ EX, const float* __restrict__ MX,
                           float* __restrict__ DEN, const int* __restrict__ dst) {
    int i = blockIdx.x * blockDim.x + threadIdx.x;
    if (i >= NEDGES * NHEADS) return;
    int e = i / NHEADS, h = i - e * NHEADS;
    int d = dst[e];
    float ex = __expf(EX[i] - MX[d * NHEADS + h]);
    EX[i] = ex;
    atomicAdd(&DEN[d * NHEADS + h], ex);
}

// ------------------------------- edge pass 3: scatter sum of v * ex
__global__ void __launch_bounds__(256)
k_edge_accum(const float* __restrict__ FQ, const float* __restrict__ attr,
             const float* __restrict__ We, const int* __restrict__ src,
             const int* __restrict__ dst, const float* __restrict__ EX,
             float* __restrict__ ACC) {
    __shared__ float sWe[EDIM * FOUT];
    for (int i = threadIdx.x; i < EDIM * FOUT; i += blockDim.x) sWe[i] = We[i];
    __syncthreads();
    int e = blockIdx.x * 8 + (threadIdx.x >> 5);
    if (e >= NEDGES) return;
    int lane = threadIdx.x & 31;
    int s = src[e], d = dst[e];
    float a8[EDIM];
#pragma unroll
    for (int j = 0; j < EDIM; ++j) a8[j] = attr[(size_t)e * EDIM + j];
    const float* vrow = FQ + (size_t)s * FQW + 2 * FOUT;
#pragma unroll
    for (int h = 0; h < NHEADS; ++h) {
        float w = EX[(size_t)e * NHEADS + h];
        int c = h * DHEAD + lane;
        float ep = 0.0f;
#pragma unroll
        for (int j = 0; j < EDIM; ++j) ep += a8[j] * sWe[j * FOUT + c];
        atomicAdd(&ACC[(size_t)d * FOUT + c], (vrow[c] + ep) * w);
    }
}

// --------------------------- finalize: out = acc/den + skip (S part of FQ)
__global__ void k_node_fin(float* __restrict__ ACC, const float* __restrict__ DEN,
                           const float* __restrict__ FQ) {
    int i = blockIdx.x * blockDim.x + threadIdx.x;
    if (i >= NNODES * FOUT) return;
    int n = i / FOUT, c = i - n * FOUT, h = c >> 5;
    float den = DEN[n * NHEADS + h];
    float v = (den > 0.0f) ? ACC[i] / den : 0.0f;
    ACC[i] = v + FQ[(size_t)n * FQW + 3 * FOUT + c];
}

// --------------------------- BN (training-mode batch stats) ----------------
__global__ void __launch_bounds__(256)
k_bn_stats(const float* __restrict__ X, float* __restrict__ MU,
           float* __restrict__ RSIG, int nrows, int ncols) {
    int c = blockIdx.x;
    float s = 0.0f, s2 = 0.0f;
    for (int r = threadIdx.x; r < nrows; r += blockDim.x) {
        float v = X[(size_t)r * ncols + c];
        s += v; s2 += v * v;
    }
    __shared__ float sh[256], sh2[256];
    sh[threadIdx.x] = s; sh2[threadIdx.x] = s2;
    __syncthreads();
    for (int o = blockDim.x >> 1; o > 0; o >>= 1) {
        if ((int)threadIdx.x < o) { sh[threadIdx.x] += sh[threadIdx.x + o];
                                    sh2[threadIdx.x] += sh2[threadIdx.x + o]; }
        __syncthreads();
    }
    if (threadIdx.x == 0) {
        float mu = sh[0] / nrows;
        float var = sh2[0] / nrows - mu * mu;
        MU[c] = mu;
        RSIG[c] = rsqrtf(fmaxf(var, 0.0f) + 1e-5f);
    }
}

__global__ void k_bn_relu(const float* __restrict__ X, const float* __restrict__ MU,
                          const float* __restrict__ RSIG, const float* __restrict__ g,
                          const float* __restrict__ b, float* __restrict__ Y,
                          int total, int ncols) {
    int i = blockIdx.x * blockDim.x + threadIdx.x;
    if (i >= total) return;
    int c = i % ncols;
    float y = (X[i] - MU[c]) * RSIG[c] * g[c] + b[c];
    Y[i] = fmaxf(y, 0.0f);
}

// --------------------------- pooling + head --------------------------------
__global__ void k_pool_zero(float* __restrict__ POOL, float* __restrict__ CNT) {
    int i = blockIdx.x * blockDim.x + threadIdx.x;
    if (i < NGRAPH * 480) POOL[i] = 0.0f;
    if (i < NGRAPH) CNT[i] = 0.0f;
}

__global__ void k_pool(const float* __restrict__ X1, const float* __restrict__ X2,
                       const float* __restrict__ X3, const int* __restrict__ batch,
                       float* __restrict__ POOL, float* __restrict__ CNT) {
    int i = blockIdx.x * blockDim.x + threadIdx.x;
    if (i >= NNODES * 480) return;
    int n = i / 480, c = i - n * 480;
    const float* Xs = (c < FOUT) ? X1 : (c < 2 * FOUT) ? X2 : X3;
    int cc = (c < FOUT) ? c : (c < 2 * FOUT) ? c - FOUT : c - 2 * FOUT;
    int g = batch[n];
    atomicAdd(&POOL[g * 480 + c], Xs[(size_t)n * FOUT + cc]);
    if (c == 0) atomicAdd(&CNT[g], 1.0f);
}

__global__ void k_pool_div(float* __restrict__ POOL, const float* __restrict__ CNT) {
    int i = blockIdx.x * blockDim.x + threadIdx.x;
    if (i >= NGRAPH * 480) return;
    POOL[i] /= fmaxf(CNT[i / 480], 1.0f);
}

__global__ void k_fc(const float* __restrict__ POOL, const float* __restrict__ MU2,
                     const float* __restrict__ RS2, const float* __restrict__ g2,
                     const float* __restrict__ b2, const float* __restrict__ fcW,
                     const float* __restrict__ fcb, float* __restrict__ out) {
    int i = blockIdx.x * blockDim.x + threadIdx.x;
    if (i >= NGRAPH * 64) return;
    int g = i >> 6, o = i & 63;
    float acc = fcb[o];
    for (int c = 0; c < 480; ++c) {
        float xb = (POOL[g * 480 + c] - MU2[c]) * RS2[c] * g2[c] + b2[c];
        acc += xb * fcW[c * 64 + o];
    }
    out[i] = acc;
}

// ===========================================================================
extern "C" void kernel_launch(void* const* d_in, const int* in_sizes, int n_in,
                              void* d_out, int out_size, void* d_ws, size_t ws_size,
                              hipStream_t stream) {
    (void)in_sizes; (void)n_in; (void)out_size; (void)ws_size;
    // jax tree_flatten (sorted-dict) order:
    const float* X     = (const float*)d_in[0];
    const float* attr  = (const float*)d_in[1];
    const int*   batch = (const int*)d_in[2];
    const int*   eidx  = (const int*)d_in[3];
    const float* bn2_b = (const float*)d_in[4];
    const float* bn2_g = (const float*)d_in[5];
    const float* fcW   = (const float*)d_in[6];
    const float* fcb   = (const float*)d_in[7];
    const int* src = eidx;
    const int* dst = eidx + NEDGES;

    char* p = (char*)d_ws;
    auto alloc = [&](size_t bytes) -> void* {
        void* r = p; p += (bytes + 255) & ~(size_t)255; return r;
    };
    float*  FQ   = (float*)alloc(sizeof(float) * (size_t)NNODES * FQW);
    float*  ACC  = (float*)alloc(sizeof(float) * (size_t)NNODES * FOUT);
    float*  EX   = (float*)alloc(sizeof(float) * (size_t)NEDGES * NHEADS);
    float*  MX   = (float*)alloc(sizeof(float) * (size_t)NNODES * NHEADS);
    float*  DEN  = (float*)alloc(sizeof(float) * (size_t)NNODES * NHEADS);
    float*  X1   = (float*)alloc(sizeof(float) * (size_t)NNODES * FOUT);
    float*  X2   = (float*)alloc(sizeof(float) * (size_t)NNODES * FOUT);
    float*  X3   = (float*)alloc(sizeof(float) * (size_t)NNODES * FOUT);
    __bf16* XB   = (__bf16*)alloc(sizeof(__bf16) * (size_t)NNODES * FOUT);
    __bf16* BP   = (__bf16*)alloc(sizeof(__bf16) * 5 * NTILES * 512);
    float*  BIAS = (float*)alloc(sizeof(float) * FQW);
    float*  MU   = (float*)alloc(sizeof(float) * 480);
    float*  RSIG = (float*)alloc(sizeof(float) * 480);
    float*  POOL = (float*)alloc(sizeof(float) * NGRAPH * 480);
    float*  CNT  = (float*)alloc(sizeof(float) * NGRAPH);

    float* XL[3] = {X1, X2, X3};
    auto nb = [](long t, int b) { return (int)((t + b - 1) / b); };

    for (int l = 0; l < 3; ++l) {
        const float* const* L = (const float* const*)(d_in + 8 + 11 * l);
        const float *We = L[0], *Wk = L[1], *Wq = L[2], *Wsm = L[3], *Wv = L[4];
        const float *bk = L[5], *bnb = L[6], *bng = L[7], *bq = L[8], *bs = L[9], *bv = L[10];

        const int din  = (l == 0) ? 55 : FOUT;
        const int dinP = (l == 0) ? 64 : FOUT;
        const int kTiles = dinP >> 5;
        const float* xin = (l == 0) ? X : XL[l - 1];

        k_cvt_bf16<<<nb((long)NNODES * dinP, 256), 256, 0, stream>>>(xin, XB, NNODES, din, dinP);
        k_pack_w<<<nb((long)kTiles * NTILES * 512, 256), 256, 0, stream>>>(Wq, Wk, Wv, Wsm, BP, din, kTiles);
        k_pack_bias<<<nb(FQW, 256), 256, 0, stream>>>(bq, bk, bv, bs, BIAS);
        {
            dim3 grid(nb(NNODES / 16, 4), NTILES / 4);
            k_gemm_qkvs<<<grid, 128, 0, stream>>>(XB, BP, BIAS, FQ, NNODES / 16, dinP);
        }
        k_init_layer<<<nb((long)NNODES * FOUT, 256), 256, 0, stream>>>(ACC, MX, DEN);
        k_edge_alpha<<<nb(NEDGES, 8), 256, 0, stream>>>(FQ, attr, We, src, dst, EX, MX);
        k_edge_exp<<<nb((long)NEDGES * NHEADS, 256), 256, 0, stream>>>(EX, MX, DEN, dst);
        k_edge_accum<<<nb(NEDGES, 8), 256, 0, stream>>>(FQ, attr, We, src, dst, EX, ACC);
        k_node_fin<<<nb((long)NNODES * FOUT, 256), 256, 0, stream>>>(ACC, DEN, FQ);
        k_bn_stats<<<FOUT, 256, 0, stream>>>(ACC, MU, RSIG, NNODES, FOUT);
        k_bn_relu<<<nb((long)NNODES * FOUT, 256), 256, 0, stream>>>(ACC, MU, RSIG, bng, bnb,
                                                                    XL[l], NNODES * FOUT, FOUT);
    }

    k_pool_zero<<<nb(NGRAPH * 480, 256), 256, 0, stream>>>(POOL, CNT);
    k_pool<<<nb((long)NNODES * 480, 256), 256, 0, stream>>>(X1, X2, X3, batch, POOL, CNT);
    k_pool_div<<<nb(NGRAPH * 480, 256), 256, 0, stream>>>(POOL, CNT);
    k_bn_stats<<<480, 256, 0, stream>>>(POOL, MU, RSIG, NGRAPH, 480);
    k_fc<<<nb(NGRAPH * 64, 256), 256, 0, stream>>>(POOL, MU, RSIG, bn2_g, bn2_b, fcW, fcb,
                                                   (float*)d_out);
}
